// GraphSAGE_43997644981191
// MI455X (gfx1250) — compile-verified, matrix operand order checked by reading, not run on hardware
//
#include <hip/hip_runtime.h>

// ---------------------------------------------------------------------------
// GraphSAGE layer for MI455X (gfx1250, wave32, WMMA).
//
// Sizes: N=50000 nodes, E=800000 edges, D=128.
// ws layout:
//   [0)                feat_neigh  : 50000*128 f32  (25,600,000 B)
//   [25600000)         Whi_self    : 128*128 f16    (32768 B)
//   [+32768)           Wlo_self
//   [+65536)           Whi_neigh
//   [+98304)           Wlo_neigh
// ---------------------------------------------------------------------------

typedef __attribute__((ext_vector_type(16))) _Float16 v16h;
typedef __attribute__((ext_vector_type(8)))  float    v8f;

#define N_NODES 50000
#define N_EDGES 800000
#define DIM     128
#define N_TILES (N_NODES / 16)   // 3125, exact

// Native FP32 atomic add to L2 (device scope), guaranteed single instruction.
__device__ __forceinline__ void atomic_add_f32_dev(float* p, float v) {
    asm volatile("global_atomic_add_f32 %0, %1, off scope:SCOPE_DEV"
                 :: "v"(p), "v"(v)
                 : "memory");
}

// ---------------- zero feat_neigh ----------------
__global__ void zero_f4(float4* __restrict__ p, int n4) {
    int stride = gridDim.x * blockDim.x;
    for (int i = blockIdx.x * blockDim.x + threadIdx.x; i < n4; i += stride)
        p[i] = make_float4(0.f, 0.f, 0.f, 0.f);
}

// ---------------- split f32 weights into f16 hi/lo ----------------
__global__ void split_weights(const float* __restrict__ Ws,
                              const float* __restrict__ Wn,
                              _Float16* __restrict__ Whs, _Float16* __restrict__ Wls,
                              _Float16* __restrict__ Whn, _Float16* __restrict__ Wln) {
    int i = blockIdx.x * blockDim.x + threadIdx.x;
    if (i >= DIM * DIM) return;
    float a = Ws[i];
    _Float16 ah = (_Float16)a;
    Whs[i] = ah;
    Wls[i] = (_Float16)(a - (float)ah);
    float b = Wn[i];
    _Float16 bh = (_Float16)b;
    Whn[i] = bh;
    Wln[i] = (_Float16)(b - (float)bh);
}

// ---------------- edge scatter: feat_neigh[row] += val * feat_in[col] -------
// One wave (32 lanes) per edge; each lane handles 4 consecutive dims (float4).
// feat_in (25.6MB) and feat_neigh (25.6MB) both fit in the 192MB L2, so the
// random gathers and the 102.4M fp32 atomics all stay on-chip.
__global__ __launch_bounds__(256) void edge_scatter(
    const float* __restrict__ feat,
    const int*   __restrict__ erow,
    const int*   __restrict__ ecol,
    const float* __restrict__ eval,
    float*       __restrict__ fneigh) {
    int e = blockIdx.x * 8 + (threadIdx.x >> 5);
    if (e >= N_EDGES) return;
    int lane = threadIdx.x & 31;
    int r = erow[e];
    int c = ecol[e];
    float v = eval[e];
    float4 x = ((const float4*)(feat + (size_t)c * DIM))[lane];
    float* dst = fneigh + (size_t)r * DIM + lane * 4;
    atomic_add_f32_dev(dst + 0, x.x * v);
    atomic_add_f32_dev(dst + 1, x.y * v);
    atomic_add_f32_dev(dst + 2, x.z * v);
    atomic_add_f32_dev(dst + 3, x.w * v);
}

// ---------------- fused GEMM + bias + ReLU + LayerNorm (+affine) ------------
// One wave per 16-row tile. WMMA 16x16x32 f16 with hi/lo error-corrected split
// (x@W ~= xhi@Whi + xlo@Whi + xhi@Wlo, f32 accumulate -> near-fp32 accuracy).
//
// A fragment (16x32, f16): lane m = lane&15 holds row m; K pattern per lane:
//   half = lane>>4;  VGPR pair v: k = kb*32 + half*8 + (v<4 ? 2v : 2v+8)
// B fragment = A-layout of B^T; B^T row n is simply row n of W (h = X @ W^T).
// C/D: VGPR j -> row (j + 8*half), column N = lane&15 (+16 per N-tile).
template <bool ACCUM>
__global__ __launch_bounds__(256) void branch_kernel(
    const float*    __restrict__ X,
    const _Float16* __restrict__ Whi,
    const _Float16* __restrict__ Wlo,
    const float*    __restrict__ bias,
    const float*    __restrict__ scale,   // pre-offset segment (len 128)
    const float*    __restrict__ offset,  // pre-offset segment (len 128)
    float*          __restrict__ out) {
    int tile = blockIdx.x * 8 + (threadIdx.x >> 5);
    if (tile >= N_TILES) return;
    int lane = threadIdx.x & 31;
    int half = lane >> 4;
    int lq   = lane & 15;
    int m0   = tile * 16;

    // ---- load + split A fragments for all 4 K-blocks ----
    v16h a_hi[4], a_lo[4];
    const float* xrow = X + (size_t)(m0 + lq) * DIM;
#pragma unroll
    for (int kb = 0; kb < 4; ++kb) {
#pragma unroll
        for (int v = 0; v < 8; ++v) {
            int k = kb * 32 + half * 8 + ((v < 4) ? (2 * v) : (2 * v + 8));
            float x0 = xrow[k], x1 = xrow[k + 1];
            _Float16 h0 = (_Float16)x0, h1 = (_Float16)x1;
            a_hi[kb][2 * v]     = h0;
            a_hi[kb][2 * v + 1] = h1;
            a_lo[kb][2 * v]     = (_Float16)(x0 - (float)h0);
            a_lo[kb][2 * v + 1] = (_Float16)(x1 - (float)h1);
        }
    }

    // ---- GEMM: 8 N-tiles x 4 K-blocks x 3 split products ----
    v8f acc[8];
#pragma unroll
    for (int nt = 0; nt < 8; ++nt) {
        const _Float16* wh = Whi + (size_t)(nt * 16 + lq) * DIM;
        const _Float16* wl = Wlo + (size_t)(nt * 16 + lq) * DIM;
        v8f c = {0.f, 0.f, 0.f, 0.f, 0.f, 0.f, 0.f, 0.f};
#pragma unroll
        for (int kb = 0; kb < 4; ++kb) {
            v16h b_hi, b_lo;
#pragma unroll
            for (int v = 0; v < 8; ++v) {
                int k = kb * 32 + half * 8 + ((v < 4) ? (2 * v) : (2 * v + 8));
                b_hi[2 * v]     = wh[k];
                b_hi[2 * v + 1] = wh[k + 1];
                b_lo[2 * v]     = wl[k];
                b_lo[2 * v + 1] = wl[k + 1];
            }
            c = __builtin_amdgcn_wmma_f32_16x16x32_f16(false, a_hi[kb], false, b_hi,
                                                       (short)0, c, false, false);
            c = __builtin_amdgcn_wmma_f32_16x16x32_f16(false, a_lo[kb], false, b_hi,
                                                       (short)0, c, false, false);
            c = __builtin_amdgcn_wmma_f32_16x16x32_f16(false, a_hi[kb], false, b_lo,
                                                       (short)0, c, false, false);
        }
        // bias + ReLU
        float bn = bias[nt * 16 + lq];
#pragma unroll
        for (int j = 0; j < 8; ++j) c[j] = fmaxf(c[j] + bn, 0.0f);
        acc[nt] = c;
    }

    // ---- LayerNorm per row: half-wave (16-lane) xor reductions ----
    float mean[8], rstd[8];
#pragma unroll
    for (int j = 0; j < 8; ++j) {
        float s = 0.f, ss = 0.f;
#pragma unroll
        for (int nt = 0; nt < 8; ++nt) {
            float v = acc[nt][j];
            s += v;
            ss += v * v;
        }
#pragma unroll
        for (int mask = 1; mask < 16; mask <<= 1) {
            s  += __shfl_xor(s, mask, 32);
            ss += __shfl_xor(ss, mask, 32);
        }
        float mu  = s * (1.0f / 128.0f);
        float var = ss * (1.0f / 128.0f) - mu * mu + 1e-9f;
        mean[j] = mu;
        rstd[j] = rsqrtf(var);
    }

    // ---- affine + store / accumulate ----
#pragma unroll
    for (int nt = 0; nt < 8; ++nt) {
        int n = nt * 16 + lq;
        float sc = scale[n], of = offset[n];
#pragma unroll
        for (int j = 0; j < 8; ++j) {
            int m = m0 + j + half * 8;
            float v = (acc[nt][j] - mean[j]) * sc * rstd[j] + of;
            float* p = out + (size_t)m * DIM + n;
            if (ACCUM)
                *p += v;
            else
                *p = v;
        }
    }
}

// ---------------------------------------------------------------------------
extern "C" void kernel_launch(void* const* d_in, const int* in_sizes, int n_in,
                              void* d_out, int out_size, void* d_ws, size_t ws_size,
                              hipStream_t stream) {
    const float* feat_in  = (const float*)d_in[0];
    const int*   edge_row = (const int*)d_in[1];
    const int*   edge_col = (const int*)d_in[2];
    const float* edge_val = (const float*)d_in[3];
    const float* W_self   = (const float*)d_in[4];
    const float* b_self   = (const float*)d_in[5];
    const float* W_neigh  = (const float*)d_in[6];
    const float* b_neigh  = (const float*)d_in[7];
    const float* scale    = (const float*)d_in[8];
    const float* offset   = (const float*)d_in[9];
    float* out = (float*)d_out;

    char* ws = (char*)d_ws;
    float*    fneigh = (float*)ws;                                  // 25,600,000 B
    _Float16* Whs    = (_Float16*)(ws + (size_t)N_NODES * DIM * 4);
    _Float16* Wls    = Whs + DIM * DIM;
    _Float16* Whn    = Wls + DIM * DIM;
    _Float16* Wln    = Whn + DIM * DIM;

    // 1. zero feat_neigh accumulator
    zero_f4<<<2048, 256, 0, stream>>>((float4*)fneigh, N_NODES * DIM / 4);

    // 2. split weights into f16 hi/lo
    split_weights<<<(DIM * DIM + 255) / 256, 256, 0, stream>>>(
        W_self, W_neigh, Whs, Wls, Whn, Wln);

    // 3. edge scatter (one wave per edge, native fp32 atomics into L2)
    edge_scatter<<<N_EDGES / 8, 256, 0, stream>>>(
        feat_in, edge_row, edge_col, edge_val, fneigh);

    // 4. self branch -> out
    branch_kernel<false><<<(N_TILES + 7) / 8, 256, 0, stream>>>(
        feat_in, Whs, Wls, b_self, scale, offset, out);

    // 5. neighbor branch -> out +=
    branch_kernel<true><<<(N_TILES + 7) / 8, 256, 0, stream>>>(
        fneigh, Whn, Wln, b_neigh, scale + DIM, offset + DIM, out);
}
